// FakeQuantLinear_89455578841487
// MI455X (gfx1250) — compile-verified, matrix operand order checked by reading, not run on hardware
//
#include <hip/hip_runtime.h>

typedef __attribute__((ext_vector_type(16))) _Float16 v16h;
typedef __attribute__((ext_vector_type(8)))  float    v8f;

#define QK_K 256

// ---------------------------------------------------------------------------
// CDNA5 async global->LDS copy (ASYNCcnt-tracked, no VGPR data movement).
// GV mode: 64-bit global VA in a VGPR pair, LDS byte offset in a VGPR.
// ---------------------------------------------------------------------------
__device__ __forceinline__ void async_load_b128(unsigned lds_off, const void* gptr) {
  asm volatile("global_load_async_to_lds_b128 %0, %1, off"
               :: "v"(lds_off), "v"((unsigned long long)(size_t)gptr)
               : "memory");
}
__device__ __forceinline__ void wait_async0() {
  asm volatile("s_wait_asynccnt 0" ::: "memory");
}

// ---------------------------------------------------------------------------
// Kernel 1: Q4_K_M fake-quantize weight (f32) -> dequantized f16 in workspace.
// One wave32 per 256-element super-block; lane holds 8 contiguous elements, so
// each 32-element sub-block is a 4-lane group.
// ---------------------------------------------------------------------------
__global__ __launch_bounds__(256) void fq_q4km_to_f16(const float* __restrict__ w,
                                                      _Float16* __restrict__ wq,
                                                      int nblocks) {
  const int gwave = (int)((blockIdx.x * 256u + threadIdx.x) >> 5);
  const int lane  = threadIdx.x & 31;
  if (gwave >= nblocks) return;

  const float* src = w + (size_t)gwave * QK_K + lane * 8;
  float4 f0 = ((const float4*)src)[0];
  float4 f1 = ((const float4*)src)[1];
  float v[8] = {f0.x, f0.y, f0.z, f0.w, f1.x, f1.y, f1.z, f1.w};

  float mn = v[0], mx = v[0];
#pragma unroll
  for (int j = 1; j < 8; ++j) { mn = fminf(mn, v[j]); mx = fmaxf(mx, v[j]); }

  // butterfly over the 4-lane group -> per-sub-block min/max
#pragma unroll
  for (int m = 1; m <= 2; m <<= 1) {
    mn = fminf(mn, __shfl_xor(mn, m, 32));
    mx = fmaxf(mx, __shfl_xor(mx, m, 32));
  }
  const float sub_min   = fminf(mn, 0.0f);
  const float sub_max   = fmaxf(mx, 0.0f);
  const float scale_raw = fmaxf(sub_max - sub_min, 1e-8f) * (1.0f / 15.0f);

  // wave-wide butterfly across the 8 sub-blocks
  float smin = scale_raw, smax = scale_raw, gmin = sub_min;
#pragma unroll
  for (int m = 4; m <= 16; m <<= 1) {
    smin = fminf(smin, __shfl_xor(smin, m, 32));
    smax = fmaxf(smax, __shfl_xor(smax, m, 32));
    gmin = fminf(gmin, __shfl_xor(gmin, m, 32));
  }

  // 6-bit quantization of the 8 sub-scales (forward value of straight-through)
  const float s_range = fmaxf(smax - smin, 1e-8f);
  const float s_int   = fminf(fmaxf(rintf((scale_raw - smin) / s_range * 63.0f), 0.0f), 63.0f);
  const float scale   = s_int * (1.0f / 63.0f) * s_range + smin;

  const float super_min = (float)(_Float16)gmin;   // f16 round-trip of super-min
  const float adj_min   = sub_min - super_min;
  const float sb        = fmaxf(scale, 1e-8f);

  _Float16* dst = wq + (size_t)gwave * QK_K + lane * 8;
#pragma unroll
  for (int j = 0; j < 8; ++j) {
    float q  = fminf(fmaxf(rintf((v[j] - super_min - adj_min) / sb), 0.0f), 15.0f);
    float dq = q * sb + adj_min + super_min;
    dst[j] = (_Float16)dq;
  }
}

// ---------------------------------------------------------------------------
// Kernel 2: one-pass x f32 -> f16 (pure bandwidth; takes conversion out of the
// GEMM hot loop and halves x's L2 traffic over the 32 tile passes).
// ---------------------------------------------------------------------------
__global__ __launch_bounds__(256) void x_to_f16(const float* __restrict__ x,
                                                _Float16* __restrict__ xh,
                                                long long n8) {
  long long i = (long long)blockIdx.x * 256 + threadIdx.x;
  if (i >= n8) return;
  const float4* s = (const float4*)x + i * 2;
  float4 a = s[0], b = s[1];
  union { _Float16 h[8]; uint4 u; } t;
  t.h[0] = (_Float16)a.x; t.h[1] = (_Float16)a.y;
  t.h[2] = (_Float16)a.z; t.h[3] = (_Float16)a.w;
  t.h[4] = (_Float16)b.x; t.h[5] = (_Float16)b.y;
  t.h[6] = (_Float16)b.z; t.h[7] = (_Float16)b.w;
  ((uint4*)xh)[i] = t.u;
}

// ---------------------------------------------------------------------------
// Kernel 3: WMMA GEMM  out[m][n] = sum_k xh[m][k] * wq[n][k] + bias[n]
// 256 threads = 8 wave32, 128x128 tile, BK=64 (two v_wmma K-steps per stage),
// double-buffered LDS filled by async global->LDS b128 copies.
// Wave grid 2x4: each wave = 64(M) x 32(N) slab = 4x2 WMMA tiles.
// ---------------------------------------------------------------------------
#define BM 128
#define BN 128
#define BK 64
#define STR 72   // halfs per LDS row: 144 B (16B-aligned, conflict-free gathers)

__global__ __launch_bounds__(256) void wmma_gemm_f16(const _Float16* __restrict__ xh,
                                                     const _Float16* __restrict__ wq,
                                                     const float* __restrict__ bias,
                                                     float* __restrict__ out,
                                                     int M, int N, int K) {
  __shared__ _Float16 As[2][BM * STR];
  __shared__ _Float16 Bs[2][BN * STR];

  const int tid  = threadIdx.x;
  const int wave = tid >> 5;
  const int lane = tid & 31;
  const int wm   = wave >> 2;     // 0..1 : 64-row (M) slab
  const int wn   = wave & 3;      // 0..3 : 32-col (N) slab
  const int r    = lane & 15;     // row within 16x16 fragment
  const int hi   = lane >> 4;     // K-half selector (adds 8 to K index)

  const int blockM = blockIdx.y * BM;
  const int blockN = blockIdx.x * BN;

  v8f acc[4][2] = {};             // 4 M-tiles x 2 N-tiles, f32 accum

  // staging: 2 threads per row, 32 halfs (64 B = 4 x b128) each
  const int lrow = tid >> 1;
  const int lk   = (tid & 1) * 32;
  const _Float16* ga = xh + (size_t)(blockM + lrow) * K + lk;
  const _Float16* gb = wq + (size_t)(blockN + lrow) * K + lk;
  const unsigned laB[2] = { (unsigned)(size_t)&As[0][lrow * STR + lk],
                            (unsigned)(size_t)&As[1][lrow * STR + lk] };
  const unsigned lbB[2] = { (unsigned)(size_t)&Bs[0][lrow * STR + lk],
                            (unsigned)(size_t)&Bs[1][lrow * STR + lk] };

  auto stage = [&](int buf, int k0) {
#pragma unroll
    for (int j = 0; j < 4; ++j) {
      async_load_b128(laB[buf] + j * 16, ga + k0 + j * 8);
      async_load_b128(lbB[buf] + j * 16, gb + k0 + j * 8);
    }
  };

  const int nstages = K / BK;
  stage(0, 0);
  wait_async0();
  __syncthreads();

  for (int s = 0; s < nstages; ++s) {
    const int buf = s & 1;
    if (s + 1 < nstages) stage(buf ^ 1, (s + 1) * BK);   // overlap with compute

    const _Float16* AsB = As[buf];
    const _Float16* BsB = Bs[buf];
#pragma unroll
    for (int kk = 0; kk < BK; kk += 32) {
      v16h afrag[4], bfrag[2];
#pragma unroll
      for (int mt = 0; mt < 4; ++mt) {
        const _Float16* base = &AsB[(wm * 64 + mt * 16 + r) * STR + kk];
        union { v16h h; unsigned int u[8]; } t;
#pragma unroll
        for (int i = 0; i < 8; ++i) {
          int k = ((i < 4) ? 2 * i : 16 + 2 * (i - 4)) + hi * 8;
          t.u[i] = *(const unsigned int*)&base[k];   // paired K halves -> one VGPR
        }
        afrag[mt] = t.h;
      }
#pragma unroll
      for (int nt = 0; nt < 2; ++nt) {
        const _Float16* base = &BsB[(wn * 32 + nt * 16 + r) * STR + kk];
        union { v16h h; unsigned int u[8]; } t;
#pragma unroll
        for (int i = 0; i < 8; ++i) {
          int k = ((i < 4) ? 2 * i : 16 + 2 * (i - 4)) + hi * 8;
          t.u[i] = *(const unsigned int*)&base[k];
        }
        bfrag[nt] = t.h;
      }
#pragma unroll
      for (int mt = 0; mt < 4; ++mt)
#pragma unroll
        for (int nt = 0; nt < 2; ++nt)
          acc[mt][nt] = __builtin_amdgcn_wmma_f32_16x16x32_f16(
              false, afrag[mt], false, bfrag[nt],
              (short)0, acc[mt][nt], false, false);
    }

    wait_async0();      // next stage's async copies landed in the other buffer
    __syncthreads();
  }

  // ---- epilogue: C/D layout -> global f32 + bias ----
#pragma unroll
  for (int mt = 0; mt < 4; ++mt) {
#pragma unroll
    for (int nt = 0; nt < 2; ++nt) {
      const int n = blockN + wn * 32 + nt * 16 + r;
      const float b = bias[n];
#pragma unroll
      for (int rr = 0; rr < 8; ++rr) {
        const int m = blockM + wm * 64 + mt * 16 + rr + 8 * hi;
        out[(size_t)m * N + n] = acc[mt][nt][rr] + b;
      }
    }
  }
}

// ---------------------------------------------------------------------------
extern "C" void kernel_launch(void* const* d_in, const int* in_sizes, int n_in,
                              void* d_out, int out_size, void* d_ws, size_t ws_size,
                              hipStream_t stream) {
  (void)n_in; (void)out_size; (void)ws_size;
  const float* x    = (const float*)d_in[0];   // (2,2048,4096) f32
  const float* w    = (const float*)d_in[1];   // (4096,4096)   f32
  const float* bias = (const float*)d_in[2];   // (4096,)       f32
  float* out        = (float*)d_out;           // (2,2048,4096) f32

  const int N = in_sizes[2];           // 4096 output features
  const int K = in_sizes[1] / N;       // 4096 reduction dim
  const int M = in_sizes[0] / K;       // 4096 = B*S

  _Float16* wq = (_Float16*)d_ws;            // 32 MB f16 dequantized weight
  _Float16* xh = wq + (size_t)N * K;         // 32 MB f16 activations

  const int nblocks = in_sizes[1] / QK_K;
  fq_q4km_to_f16<<<(nblocks * 32 + 255) / 256, 256, 0, stream>>>(w, wq, nblocks);

  const long long n8 = (long long)M * K / 8;
  x_to_f16<<<(unsigned)((n8 + 255) / 256), 256, 0, stream>>>(x, xh, n8);

  dim3 grid(N / BN, M / BM);
  wmma_gemm_f16<<<grid, 256, 0, stream>>>(xh, wq, bias, out, M, N, K);
}
